// SelfAttentionGNNFusion_14508399526292
// MI455X (gfx1250) — matrix operand order repaired and checked by reading, not compiled, not run on hardware
//
#include <hip/hip_runtime.h>
#include <hip/hip_bf16.h>

typedef _Float16 h16;
typedef __attribute__((ext_vector_type(16))) _Float16 v16h;
typedef __attribute__((ext_vector_type(8)))  _Float16 v8h;
typedef __attribute__((ext_vector_type(8)))  float    v8f;
typedef __attribute__((ext_vector_type(4)))  int      i4;

#define S_LEN 2048
#define EMB   512
#define E3    1536
#define HD    128
#define NB    8

// ---- CDNA5 async global->LDS copy (ASYNCcnt path), with safe fallback ----
#if defined(__has_builtin)
#if __has_builtin(__builtin_amdgcn_global_load_async_to_lds_b128) && \
    __has_builtin(__builtin_amdgcn_s_wait_asynccnt)
#define USE_ASYNC_LDS 1
#endif
#endif
#ifndef USE_ASYNC_LDS
#define USE_ASYNC_LDS 0
#endif

__device__ __forceinline__ void copy16_g2l(const h16* __restrict__ g,
                                           h16* __restrict__ l) {
#if USE_ASYNC_LDS
  __builtin_amdgcn_global_load_async_to_lds_b128(
      (__attribute__((address_space(1))) i4*)g,
      (__attribute__((address_space(3))) i4*)l, 0, 0);
#else
  *(v8h*)l = *(const v8h*)g;
#endif
}

__device__ __forceinline__ void async_stage_wait() {
#if USE_ASYNC_LDS
  __builtin_amdgcn_s_wait_asynccnt(0);
#endif
}

__device__ __forceinline__ v8f wmma16x16x32(v16h a, v16h b, v8f c) {
  return __builtin_amdgcn_wmma_f32_16x16x32_f16(false, a, false, b, (short)0, c,
                                                false, false);
}

// ---------------- f32 -> f16 convert ----------------
__global__ void cvt_f32_to_f16(const float* __restrict__ src,
                               h16* __restrict__ dst, int n) {
  int i = blockIdx.x * blockDim.x + threadIdx.x;
  if (i < n) dst[i] = (h16)src[i];
}

// ---------------- zero fill ----------------
__global__ void zero_f32(float* __restrict__ p, int n) {
  int i = blockIdx.x * blockDim.x + threadIdx.x;
  if (i < n) p[i] = 0.0f;
}

// ---------------- WMMA GEMM with LDS-staged, double-buffered B panel -------
// C[M,N] = A[M,K] @ B^T, B stored row-major [N,K] (PyTorch Linear weight).
// 128 threads = 4 waves; block tile 128x64; wave tile 32x64 (2x4 WMMA frags).
// B panel (64 cols x 32 k) staged in LDS per k-step, double buffered, async.
__global__ __launch_bounds__(128)
void gemm_wmma(const h16* __restrict__ A, int lda,
               const h16* __restrict__ B, int ldb, int K,
               const float* __restrict__ bias,
               float* __restrict__ Cf, int ldcf,
               h16* __restrict__ Ch, int ldch,
               float* __restrict__ meanAcc, int rowsPerBatch, float meanScale) {
  __shared__ __align__(32) h16 Bs[2][64][32];

  const int tid  = threadIdx.x;
  const int lane = tid & 31;
  const int wave = tid >> 5;
  const int hl   = lane >> 4;
  const int mr   = lane & 15;
  const int m0   = blockIdx.y * 128 + wave * 32;
  const int n0   = blockIdx.x * 64;

  // staging map: thread t owns Bs[.][t>>1][(t&1)*16 .. +15]  (two b128 each)
  const int sr = tid >> 1;
  const int sc = (tid & 1) * 16;
  const h16* bsrc = B + (size_t)(n0 + sr) * ldb + sc;

  v8f zero = {};
  v8f acc[2][4];
  for (int mi = 0; mi < 2; ++mi)
    for (int t = 0; t < 4; ++t) acc[mi][t] = zero;

  const h16* arow0 = A + (size_t)(m0 + mr) * lda;
  const h16* arow1 = A + (size_t)(m0 + 16 + mr) * lda;

  // prologue: stage k0 = 0 into buffer 0
  copy16_g2l(bsrc,     &Bs[0][sr][sc]);
  copy16_g2l(bsrc + 8, &Bs[0][sr][sc + 8]);

  int cur = 0;
  for (int k0 = 0; k0 < K; k0 += 32) {
    async_stage_wait();   // current buffer's async copies done (this wave)
    __syncthreads();      // all waves staged + prior reads of other buf done
    if (k0 + 32 < K) {    // kick off next k-step into the other buffer
      const h16* bn = bsrc + k0 + 32;
      copy16_g2l(bn,     &Bs[cur ^ 1][sr][sc]);
      copy16_g2l(bn + 8, &Bs[cur ^ 1][sr][sc + 8]);
    }
    // A fragments (16x32, contiguous k = 16*hl + s), two M sub-tiles
    v16h a0, a1;
    {
      v8h lo = *(const v8h*)(arow0 + k0 + hl * 16);
      v8h hi = *(const v8h*)(arow0 + k0 + hl * 16 + 8);
      for (int i = 0; i < 8; ++i) { a0[i] = lo[i]; a0[8 + i] = hi[i]; }
    }
    {
      v8h lo = *(const v8h*)(arow1 + k0 + hl * 16);
      v8h hi = *(const v8h*)(arow1 + k0 + hl * 16 + 8);
      for (int i = 0; i < 8; ++i) { a1[i] = lo[i]; a1[8 + i] = hi[i]; }
    }
    for (int t = 0; t < 4; ++t) {
      v16h bf = *(const v16h*)&Bs[cur][t * 16 + mr][hl * 16];
      acc[0][t] = wmma16x16x32(a0, bf, acc[0][t]);
      acc[1][t] = wmma16x16x32(a1, bf, acc[1][t]);
    }
    cur ^= 1;
  }

  for (int mi = 0; mi < 2; ++mi) {
    for (int t = 0; t < 4; ++t) {
      int col = n0 + t * 16 + mr;
      float bv = bias ? bias[col] : 0.0f;
      for (int r = 0; r < 8; ++r) {
        int row = m0 + mi * 16 + r + hl * 8;
        float v = acc[mi][t][r] + bv;
        if (Cf) Cf[(size_t)row * ldcf + col] = v;
        if (Ch) Ch[(size_t)row * ldch + col] = (h16)v;
        if (meanAcc)
          atomicAdd(meanAcc + (size_t)(row / rowsPerBatch) * 512 + col,
                    v * meanScale);
      }
    }
  }
}

// ---------------- flash attention ----------------
__global__ __launch_bounds__(128)
void attention_flash(const h16* __restrict__ qkv, h16* __restrict__ ctx) {
  __shared__ __align__(32) h16 Kc[32][HD];      // K chunk  [j][d]
  __shared__ __align__(32) h16 Vt[HD][32];      // V chunk transposed [d][j]
  __shared__ __align__(32) h16 Pbuf[4][16][32]; // per-wave probs tile

  const int tid  = threadIdx.x;
  const int lane = tid & 31;
  const int wave = tid >> 5;
  const int hl   = lane >> 4;
  const int mr   = lane & 15;

  const int blk  = blockIdx.x;
  const int qblk = blk & 31;
  const int h    = (blk >> 5) & 3;
  const int b    = blk >> 7;

  const size_t base = (size_t)b * S_LEN * E3 + (size_t)h * HD;
  const h16* Qp = qkv + base;
  const h16* Kp = qkv + base + EMB;
  const h16* Vp = qkv + base + 2 * EMB;

  const int s0 = qblk * 64 + wave * 16;

  v16h aq[4];
  {
    const h16* qrow = Qp + (size_t)(s0 + mr) * E3;
    for (int kk = 0; kk < 4; ++kk) {
      int k0 = kk * 32;
      v8h lo = *(const v8h*)(qrow + k0 + hl * 16);
      v8h hi = *(const v8h*)(qrow + k0 + hl * 16 + 8);
      for (int i = 0; i < 8; ++i) { aq[kk][i] = lo[i]; aq[kk][8 + i] = hi[i]; }
    }
  }

  v8f zero = {};
  v8f acc[8];
  for (int t = 0; t < 8; ++t) acc[t] = zero;
  float rmax[8], rsum[8];
  for (int r = 0; r < 8; ++r) { rmax[r] = -1e30f; rsum[r] = 0.0f; }

  const float scale = 0.08838834764831845f;  // 1/sqrt(128)

  for (int j0 = 0; j0 < S_LEN; j0 += 32) {
    // ---- stage K chunk (async) and V^T chunk (transpose scatter) ----
    {
      int jr = tid & 31;
      int qd = tid >> 5;
      const h16* krow = Kp + (size_t)(j0 + jr) * E3 + qd * 32;
      const h16* vrow = Vp + (size_t)(j0 + jr) * E3 + qd * 32;
      for (int u = 0; u < 4; ++u) {
        copy16_g2l(krow + u * 8, &Kc[jr][qd * 32 + u * 8]);
        v8h vv = *(const v8h*)(vrow + u * 8);
        for (int i = 0; i < 8; ++i) Vt[qd * 32 + u * 8 + i][jr] = vv[i];
      }
    }
    async_stage_wait();
    __syncthreads();

    // ---- scores via WMMA ----
    v8f sc0 = zero, sc1 = zero;
    for (int kk = 0; kk < 4; ++kk) {
      int k0 = kk * 32;
      v16h b0 = *(const v16h*)&Kc[mr][k0 + hl * 16];
      v16h b1 = *(const v16h*)&Kc[16 + mr][k0 + hl * 16];
      sc0 = wmma16x16x32(aq[kk], b0, sc0);
      sc1 = wmma16x16x32(aq[kk], b1, sc1);
    }

    // ---- online softmax (stats in C-fragment layout) ----
    for (int r = 0; r < 8; ++r) {
      float x0 = sc0[r] * scale, x1 = sc1[r] * scale;
      float mx = fmaxf(x0, x1);
      for (int off = 1; off < 16; off <<= 1)
        mx = fmaxf(mx, __shfl_xor(mx, off, 32));
      float mnew = fmaxf(rmax[r], mx);
      float corr = __expf(rmax[r] - mnew);
      float p0 = __expf(x0 - mnew);
      float p1 = __expf(x1 - mnew);
      float ps = p0 + p1;
      for (int off = 1; off < 16; off <<= 1) ps += __shfl_xor(ps, off, 32);
      rsum[r] = rsum[r] * corr + ps;
      rmax[r] = mnew;
      for (int t = 0; t < 8; ++t) acc[t][r] *= corr;
      int prow = r + hl * 8;
      Pbuf[wave][prow][mr]      = (h16)p0;
      Pbuf[wave][prow][16 + mr] = (h16)p1;
    }
    __syncthreads();

    // ---- P @ V ----
    v16h ap;
    {
      v8h lo = *(const v8h*)&Pbuf[wave][mr][hl * 16];
      v8h hi = *(const v8h*)&Pbuf[wave][mr][hl * 16 + 8];
      for (int i = 0; i < 8; ++i) { ap[i] = lo[i]; ap[8 + i] = hi[i]; }
    }
    for (int t = 0; t < 8; ++t) {
      v16h bv = *(const v16h*)&Vt[t * 16 + mr][hl * 16];
      acc[t] = wmma16x16x32(ap, bv, acc[t]);
    }
    __syncthreads();
  }

  for (int r = 0; r < 8; ++r) {
    float inv = 1.0f / rsum[r];
    int row = s0 + r + hl * 8;
    h16* orow = ctx + ((size_t)b * S_LEN + row) * EMB + h * HD;
    for (int t = 0; t < 8; ++t)
      orow[t * 16 + mr] = (h16)(acc[t][r] * inv);
  }
}

// ---------------- banded distance-weighted aggregation ----------------
__global__ void banded_agg(const float* __restrict__ attn,
                           h16* __restrict__ node) {
  int idx = blockIdx.x * blockDim.x + threadIdx.x;
  if (idx >= NB * S_LEN * EMB) return;
  int d = idx & (EMB - 1);
  int i = (idx >> 9) & (S_LEN - 1);
  int b = idx >> 20;
  const float* col = attn + (size_t)b * S_LEN * EMB + d;
  float wsum = 0.0f, accv = 0.0f;
  for (int off = 1; off <= 10; ++off) {
    float w = __expf(-0.5f * (float)off);
    int jm = i - off, jp = i + off;
    if (jm >= 0)    { wsum += w; accv += w * col[(size_t)jm * EMB]; }
    if (jp < S_LEN) { wsum += w; accv += w * col[(size_t)jp * EMB]; }
  }
  float v = accv / (wsum + 1e-8f);
  node[((size_t)b * S_LEN + i) * 1024 + EMB + d] = (h16)v;
}

// ---------------- final projection ----------------
__global__ void final_proj(const float* __restrict__ mean,
                           const float* __restrict__ fpw,
                           const float* __restrict__ fpb,
                           float* __restrict__ out) {
  int idx = blockIdx.x * blockDim.x + threadIdx.x;
  if (idx >= NB * EMB) return;
  int n = idx & (EMB - 1);
  int b = idx >> 9;
  const float* mrow = mean + b * EMB;
  const float* wrow = fpw + (size_t)n * EMB;
  float s = fpb[n];
  for (int k = 0; k < EMB; ++k) s += mrow[k] * wrow[k];
  out[idx] = s;
}

extern "C" void kernel_launch(void* const* d_in, const int* in_sizes, int n_in,
                              void* d_out, int out_size, void* d_ws,
                              size_t ws_size, hipStream_t stream) {
  const float* x    = (const float*)d_in[0];
  const float* inw  = (const float*)d_in[1];
  const float* inb  = (const float*)d_in[2];
  const float* outw = (const float*)d_in[3];
  const float* outb = (const float*)d_in[4];
  const float* mpw  = (const float*)d_in[5];
  const float* mpb  = (const float*)d_in[6];
  const float* fpw  = (const float*)d_in[7];
  const float* fpb  = (const float*)d_in[8];
  float* out = (float*)d_out;

  char* ws = (char*)d_ws;
  size_t off = 0;
  auto alloc = [&](size_t bytes) -> void* {
    void* p = ws + off;
    off += (bytes + 255) & ~(size_t)255;
    return p;
  };

  const int nX    = NB * S_LEN * EMB;
  const int nQKV  = NB * S_LEN * E3;
  const int nInW  = E3 * EMB;
  const int nOutW = EMB * EMB;
  const int nMpW  = EMB * 1024;

  h16*   xh      = (h16*)alloc((size_t)nX * 2);
  h16*   w_inh   = (h16*)alloc((size_t)nInW * 2);
  h16*   w_outh  = (h16*)alloc((size_t)nOutW * 2);
  h16*   w_mph   = (h16*)alloc((size_t)nMpW * 2);
  h16*   qkvh    = (h16*)alloc((size_t)nQKV * 2);
  h16*   ctxh    = (h16*)alloc((size_t)nX * 2);
  float* attnf   = (float*)alloc((size_t)nX * 4);
  h16*   node    = (h16*)alloc((size_t)NB * S_LEN * 1024 * 2);
  float* meanacc = (float*)alloc((size_t)NB * EMB * 4);

  cvt_f32_to_f16<<<(nX + 255) / 256, 256, 0, stream>>>(x, xh, nX);
  cvt_f32_to_f16<<<(nInW + 255) / 256, 256, 0, stream>>>(inw, w_inh, nInW);
  cvt_f32_to_f16<<<(nOutW + 255) / 256, 256, 0, stream>>>(outw, w_outh, nOutW);
  cvt_f32_to_f16<<<(nMpW + 255) / 256, 256, 0, stream>>>(mpw, w_mph, nMpW);

  // QKV projection: [16384,512] x [512,1536]
  gemm_wmma<<<dim3(E3 / 64, (NB * S_LEN) / 128), 128, 0, stream>>>(
      xh, EMB, w_inh, EMB, EMB, inb, nullptr, 0, qkvh, E3, nullptr, 0, 0.0f);

  // flash attention
  attention_flash<<<NB * 4 * (S_LEN / 64), 128, 0, stream>>>(qkvh, ctxh);

  // out_proj: f32 for banded agg + f16 into node[:, :512]
  gemm_wmma<<<dim3(EMB / 64, (NB * S_LEN) / 128), 128, 0, stream>>>(
      ctxh, EMB, w_outh, EMB, EMB, outb, attnf, EMB, node, 1024, nullptr, 0,
      0.0f);

  zero_f32<<<(NB * EMB + 255) / 256, 256, 0, stream>>>(meanacc, NB * EMB);
  banded_agg<<<(NB * S_LEN * EMB + 255) / 256, 256, 0, stream>>>(attnf, node);

  // message-passing GEMM with fused mean over sequence
  gemm_wmma<<<dim3(EMB / 64, (NB * S_LEN) / 128), 128, 0, stream>>>(
      node, 1024, w_mph, 1024, 1024, mpb, nullptr, 0, nullptr, 0, meanacc,
      S_LEN, 1.0f / (float)S_LEN);

  final_proj<<<(NB * EMB + 255) / 256, 256, 0, stream>>>(meanacc, fpw, fpb,
                                                         out);
}
